// AttentionBlock_36756330119401
// MI455X (gfx1250) — matrix operand order repaired and checked by reading, not compiled, not run on hardware
//
#include <hip/hip_runtime.h>

// ---------------------------------------------------------------------------
// AttentionBlock for MI455X (gfx1250): bf16 WMMA + TDM weight staging.
//   x:[32,64,64,256] f32 ; per (b,h): seq=64 (w axis), ch=256.
//   out = (softmax(QK^T * s) V) Wo + bo + x
// ---------------------------------------------------------------------------

typedef __attribute__((ext_vector_type(4)))  unsigned short v4u;
typedef __attribute__((ext_vector_type(8)))  unsigned short v8u;
typedef __attribute__((ext_vector_type(16))) unsigned short v16u;
typedef __attribute__((ext_vector_type(16))) __bf16        v16bf;
typedef __attribute__((ext_vector_type(8)))  float          v8f;
typedef __attribute__((ext_vector_type(4)))  unsigned       v4ui;
typedef __attribute__((ext_vector_type(8)))  int            v8i;
typedef __attribute__((ext_vector_type(4)))  int            v4i;

#define WSEQ 64
#define CCH  256
#define XPAD 264   // halves per row, [64][256] bf16 tiles (breaks 512B-stride bank conflicts)
#define VPAD 72    // halves per row, V^T [256][64] bf16
#define SPAD 68    // floats per row, scores [64][64] f32
#define PPAD 72    // halves per row, P [64][64] bf16
#define WBROW 264  // halves per staged weight row (TDM pad: 512B + 16B)
#define WBSTRIDE (128 * WBROW)            // halves per staged half-matrix (33792)

#define OFF_XT 0
#define OFF_QT (OFF_XT + WSEQ*XPAD*2)     // 33792
#define OFF_KT (OFF_QT + WSEQ*XPAD*2)     // 67584
#define OFF_VT (OFF_KT + WSEQ*XPAD*2)     // 101376
#define OFF_S  (OFF_VT + CCH*VPAD*2)      // 138240
#define OFF_P  (OFF_S  + WSEQ*SPAD*4)     // 155648
#define OFF_WB (OFF_P  + WSEQ*PPAD*2)     // 164864  (double-buffered TDM weight stage)
#define LDS_BYTES (OFF_WB + 2*WBSTRIDE*2) // 300032 (< 320 KB/WGP)

// combined scale: 1/(c*sqrt(0.5)) applied, then softmax(./sqrt(c))
#define SCORE_SCALE 3.4526698e-4f   // sqrt(2)/256^1.5

__device__ __forceinline__ unsigned short f2bfu(float f) {
  unsigned u = __builtin_bit_cast(unsigned, f);
  u += 0x7FFFu + ((u >> 16) & 1u);           // round-to-nearest-even
  return (unsigned short)(u >> 16);
}

// A fragment 16x32 bf16 from row-major [rows][stride] halves, tile (mo,ko).
__device__ __forceinline__ v16bf ldA(const unsigned short* base, int stride, int mo, int ko) {
  const int lane = threadIdx.x & 31;
  const int lh = lane & 15, hi = lane >> 4;
  const unsigned short* p = base + (mo * 16 + lh) * stride + ko * 32 + 8 * hi;
  v8u a = *(const v8u*)p;
  v8u b = *(const v8u*)(p + 16);
  v16u r = __builtin_shufflevector(a, b, 0,1,2,3,4,5,6,7,8,9,10,11,12,13,14,15);
  return __builtin_bit_cast(v16bf, r);
}

// B fragment 32x16 bf16 from [n][k] layout: lane L: N=L&15, k = 8*(L>>4)+0..7, 16+8*(L>>4)+0..7.
__device__ __forceinline__ v16bf ldB(const unsigned short* base, int stride, int no, int ko) {
  const int lane = threadIdx.x & 31;
  const int lh = lane & 15, hi = lane >> 4;
  const unsigned short* p = base + (no * 16 + lh) * stride + ko * 32 + 8 * hi;
  v8u a = *(const v8u*)p;
  v8u b = *(const v8u*)(p + 16);
  v16u r = __builtin_shufflevector(a, b, 0,1,2,3,4,5,6,7,8,9,10,11,12,13,14,15);
  return __builtin_bit_cast(v16bf, r);
}

__device__ __forceinline__ v8f wmma_bf16(v16bf a, v16bf b, v8f c) {
  return __builtin_amdgcn_wmma_f32_16x16x32_bf16(false, a, false, b, (short)0, c, false, false);
}

// TDM: DMA one 128x256 bf16 half-matrix (row padded 512B->528B) into LDS.
// D# per CDNA5 ISA: group0 {count,lds,global,type},
// group1 {flags, dims, tile, stride}; groups 2/3 zero (2-D tensor).
__device__ __forceinline__ void tdm_load_chunk(const unsigned short* src, unsigned lds_off) {
  unsigned long long ga = (unsigned long long)(const void*)src;
  v4ui g0;
  g0[0] = 1u;                                                  // count=1 (valid), user mode
  g0[1] = lds_off;                                             // lds_addr (bytes)
  g0[2] = (unsigned)ga;                                        // global_addr[31:0]
  g0[3] = (unsigned)((ga >> 32) & 0x01FFFFFFu) | (2u << 30);   // global_addr[56:32] | type=2
  v8i g1;
  g1[0] = (int)((1u << 16)        // data_size = 2 bytes
              | (1u << 20)        // pad_enable
              | (6u << 22)        // pad_interval: 128 dwords (one 512B row)
              | (3u << 25));      // pad_amount: 4 dwords (16B) -> 264-half stride
  g1[1] = (int)(256u << 16);      // tensor_dim0 = 256 elements (low16)
  g1[2] = (int)(128u << 16);      // tensor_dim1 = 128 rows (low16); dim0 hi16 = 0
  g1[3] = (int)(256u << 16);      // tile_dim0 = 256; dim1 hi16 = 0
  g1[4] = 128;                    // tile_dim1 = 128 rows
  g1[5] = 256;                    // tensor_dim0_stride = 256 elements
  g1[6] = 0;
  g1[7] = 0;
  v4i z = {0, 0, 0, 0};
#if __clang_major__ >= 23
  v8i z8 = {0, 0, 0, 0, 0, 0, 0, 0};
  __builtin_amdgcn_tensor_load_to_lds(g0, g1, z, z, z8, 0);
#else
  __builtin_amdgcn_tensor_load_to_lds(g0, g1, z, z, 0);
#endif
}

// ---------------------------------------------------------------------------
// Kernel 0: W[c][d] f32 -> WT[d][c] bf16 into workspace (B-fragment friendly).
// Chunk c (0..7) = matrix c>>1, rows 128*(c&1).. : contiguous at ws + c*32768.
// ---------------------------------------------------------------------------
__global__ __launch_bounds__(256) void prep_weights(
    const float* __restrict__ Wq, const float* __restrict__ Wk,
    const float* __restrict__ Wv, const float* __restrict__ Wo,
    unsigned short* __restrict__ ws) {
  const int i = blockIdx.x * 256 + threadIdx.x;     // 0..65535
  const int sel = blockIdx.y;                       // 0..3
  const float* W = (sel == 0) ? Wq : (sel == 1) ? Wk : (sel == 2) ? Wv : Wo;
  const int n = i >> 8, k = i & 255;
  ws[sel * 65536 + n * 256 + k] = f2bfu(W[k * 256 + n]);
}

// ---------------------------------------------------------------------------
// Kernel 1: one workgroup per (b,h). 8 wave32 per WG.
// ---------------------------------------------------------------------------
__global__ __launch_bounds__(256) void attn_kernel(
    const float* __restrict__ x,
    const float* __restrict__ bq, const float* __restrict__ bk,
    const float* __restrict__ bv, const float* __restrict__ bo,
    const unsigned short* __restrict__ wsW,
    float* __restrict__ out) {
  extern __shared__ char smem[];
  unsigned short* XT = (unsigned short*)(smem + OFF_XT);   // [64][XPAD] bf16 x tile
  unsigned short* QT = (unsigned short*)(smem + OFF_QT);   // [64][XPAD] (reused as O_attn)
  unsigned short* KT = (unsigned short*)(smem + OFF_KT);   // [64][XPAD]
  unsigned short* VT = (unsigned short*)(smem + OFF_VT);   // [256][VPAD] V^T ([ch][row])
  float*          SM = (float*)(smem + OFF_S);             // [64][SPAD] f32 scores
  unsigned short* PT = (unsigned short*)(smem + OFF_P);    // [64][PPAD] bf16 probs
  unsigned short* WBbase = (unsigned short*)(smem + OFF_WB);

  const int bh   = blockIdx.x;                  // 0..2047
  const int lane = threadIdx.x & 31;
  const int wave = threadIdx.x >> 5;            // 0..7
  const int lh = lane & 15, hi = lane >> 4;
  const int mo = wave & 3;                      // M-tile owned by this wave
  const int g  = wave >> 2;                     // N-group owned by this wave
  const long base_row = (long)bh * WSEQ;

  // Kick off TDM of weight chunk 0 before touching x (overlaps phase 0).
  if (wave == 0) tdm_load_chunk(wsW, OFF_WB);

  // ---- Phase 0: x tile f32 -> bf16 in LDS -------------------------------
  {
    const float4* xg = (const float4*)(x + base_row * CCH);
    for (int i = threadIdx.x; i < WSEQ * CCH / 4; i += 256) {
      float4 v = xg[i];
      const int row = i >> 6;                   // 64 float4 per row
      const int ch  = (i & 63) * 4;
      v4u pk; pk[0] = f2bfu(v.x); pk[1] = f2bfu(v.y); pk[2] = f2bfu(v.z); pk[3] = f2bfu(v.w);
      *(v4u*)(XT + row * XPAD + ch) = pk;
    }
  }

  // ---- Phase 1: Q/K/V = X * W + b, weights streamed by TDM --------------
  // chunk c: matrix sel=c>>1 (q,k,v), rows 128*(c&1).. ; double-buffered.
  for (int c = 0; c < 6; ++c) {
    if (wave == 0) __builtin_amdgcn_s_wait_tensorcnt(0);   // chunk c landed
    __syncthreads();                                       // data visible; prev buffer free
    if (wave == 0 && c < 5)
      tdm_load_chunk(wsW + (c + 1) * 32768,
                     OFF_WB + (unsigned)(((c + 1) & 1) * (WBSTRIDE * 2)));
    const int sel = c >> 1, halfc = c & 1;
    const unsigned short* WB = WBbase + (c & 1) * WBSTRIDE;
    const float* bias = (sel == 0) ? bq : (sel == 1) ? bk : bv;
#pragma unroll
    for (int pr = 0; pr < 2; ++pr) {             // two dual-accumulator pairs
      const int nl0 = g * 4 + pr * 2;            // local n-tile 0..7 within chunk
      v8f acc0 = {}, acc1 = {};
#pragma unroll
      for (int ko = 0; ko < 8; ++ko) {
        v16bf a = ldA(XT, XPAD, mo, ko);         // shared A, 2 independent chains
        acc0 = wmma_bf16(a, ldB(WB, WBROW, nl0,     ko), acc0);
        acc1 = wmma_bf16(a, ldB(WB, WBROW, nl0 + 1, ko), acc1);
      }
      const int n0 = (halfc * 8 + nl0) * 16 + lh, n1 = n0 + 16;
      const float b0 = bias[n0], b1 = bias[n1];
      if (sel < 2) {
        unsigned short* dst = (sel == 0) ? QT : KT;
#pragma unroll
        for (int p = 0; p < 8; ++p) {
          const int r = (mo * 16 + 8 * hi + p) * XPAD;
          dst[r + n0] = f2bfu(acc0[p] + b0);
          dst[r + n1] = f2bfu(acc1[p] + b1);
        }
      } else {                                   // V stored transposed: [ch][row]
        v8u pk0, pk1;
#pragma unroll
        for (int p = 0; p < 8; ++p) { pk0[p] = f2bfu(acc0[p] + b0); pk1[p] = f2bfu(acc1[p] + b1); }
        *(v8u*)(VT + n0 * VPAD + mo * 16 + 8 * hi) = pk0;
        *(v8u*)(VT + n1 * VPAD + mo * 16 + 8 * hi) = pk1;
      }
    }
  }
  __syncthreads();

  // Prefetch both Wo half-matrices via TDM; they land during phases 2-3.
  if (wave == 0) {
    tdm_load_chunk(wsW + 6 * 32768, OFF_WB);
    tdm_load_chunk(wsW + 7 * 32768, OFF_WB + (unsigned)(WBSTRIDE * 2));
  }

  // ---- Phase 2: S = Q K^T * scale (two tiles/wave, shared A) ------------
  {
    const int jo = g;                            // tiles (mo,jo) and (mo,jo+2)
    v8f acc0 = {}, acc1 = {};
#pragma unroll
    for (int ko = 0; ko < 8; ++ko) {
      v16bf a = ldA(QT, XPAD, mo, ko);
      acc0 = wmma_bf16(a, ldB(KT, XPAD, jo,     ko), acc0);
      acc1 = wmma_bf16(a, ldB(KT, XPAD, jo + 2, ko), acc1);
    }
    const int j0 = jo * 16 + lh, j1 = j0 + 32;
#pragma unroll
    for (int p = 0; p < 8; ++p) {
      const int r = (mo * 16 + 8 * hi + p) * SPAD;
      SM[r + j0] = acc0[p] * SCORE_SCALE;
      SM[r + j1] = acc1[p] * SCORE_SCALE;
    }
  }
  __syncthreads();

  // ---- Softmax over j (one thread per row) ------------------------------
  if (threadIdx.x < WSEQ) {
    float* row = SM + threadIdx.x * SPAD;
    float mx = row[0];
    for (int j = 1; j < WSEQ; ++j) mx = fmaxf(mx, row[j]);
    float sum = 0.f;
    for (int j = 0; j < WSEQ; ++j) { float e = __expf(row[j] - mx); row[j] = e; sum += e; }
    const float inv = 1.0f / sum;
    unsigned short* prow = PT + threadIdx.x * PPAD;
    for (int j = 0; j < WSEQ; ++j) prow[j] = f2bfu(row[j] * inv);
  }
  __syncthreads();

  // ---- Phase 3: O_attn = P V (K-dim 64) -> reuse QT ---------------------
  {
#pragma unroll
    for (int pr = 0; pr < 4; ++pr) {
      const int no0 = g + 2 * pr, no1 = no0 + 8;
      v8f acc0 = {}, acc1 = {};
#pragma unroll
      for (int ko = 0; ko < 2; ++ko) {
        v16bf a = ldA(PT, PPAD, mo, ko);
        acc0 = wmma_bf16(a, ldB(VT, VPAD, no0, ko), acc0);
        acc1 = wmma_bf16(a, ldB(VT, VPAD, no1, ko), acc1);
      }
      const int n0 = no0 * 16 + lh, n1 = n0 + 128;
#pragma unroll
      for (int p = 0; p < 8; ++p) {
        const int r = (mo * 16 + 8 * hi + p) * XPAD;
        QT[r + n0] = f2bfu(acc0[p]);
        QT[r + n1] = f2bfu(acc1[p]);
      }
    }
  }
  if (wave == 0) __builtin_amdgcn_s_wait_tensorcnt(0);     // Wo chunks landed
  __syncthreads();

  // ---- Phase 4: out = O_attn * Wo + bo + x (Wo from TDM stage) ----------
  {
    const unsigned short* WB = WBbase + g * WBSTRIDE;      // buf g = rows 128*g..
#pragma unroll
    for (int pr = 0; pr < 4; ++pr) {
      const int nl0 = pr * 2;
      v8f acc0 = {}, acc1 = {};
#pragma unroll
      for (int ko = 0; ko < 8; ++ko) {
        v16bf a = ldA(QT, XPAD, mo, ko);
        acc0 = wmma_bf16(a, ldB(WB, WBROW, nl0,     ko), acc0);
        acc1 = wmma_bf16(a, ldB(WB, WBROW, nl0 + 1, ko), acc1);
      }
      const int n0 = (g * 8 + nl0) * 16 + lh, n1 = n0 + 16;
      const float bb0 = bo[n0], bb1 = bo[n1];
#pragma unroll
      for (int p = 0; p < 8; ++p) {
        const long r = (base_row + mo * 16 + 8 * hi + p) * CCH;
        out[r + n0] = acc0[p] + bb0 + x[r + n0];
        out[r + n1] = acc1[p] + bb1 + x[r + n1];
      }
    }
  }
}

// ---------------------------------------------------------------------------
extern "C" void kernel_launch(void* const* d_in, const int* in_sizes, int n_in,
                              void* d_out, int out_size, void* d_ws, size_t ws_size,
                              hipStream_t stream) {
  const float* x  = (const float*)d_in[0];
  const float* Wq = (const float*)d_in[1];
  const float* bq = (const float*)d_in[2];
  const float* Wk = (const float*)d_in[3];
  const float* bk = (const float*)d_in[4];
  const float* Wv = (const float*)d_in[5];
  const float* bv = (const float*)d_in[6];
  const float* Wo = (const float*)d_in[7];
  const float* bo = (const float*)d_in[8];
  unsigned short* ws = (unsigned short*)d_ws;   // 512 KB of bf16 transposed weights
  float* out = (float*)d_out;

  (void)hipFuncSetAttribute((const void*)attn_kernel,
                            hipFuncAttributeMaxDynamicSharedMemorySize, LDS_BYTES);

  prep_weights<<<dim3(256, 4), 256, 0, stream>>>(Wq, Wk, Wv, Wo, ws);
  attn_kernel<<<dim3(32 * 64), 256, LDS_BYTES, stream>>>(x, bq, bk, bv, bo, ws, out);
}
